// SelfAttention_56547539419709
// MI455X (gfx1250) — matrix operand order repaired and checked by reading, not compiled, not run on hardware
//
#include <hip/hip_runtime.h>
#include <hip/hip_bf16.h>
#include <stdint.h>

typedef __bf16 bf16;
typedef __attribute__((ext_vector_type(16))) bf16 v16bf;
typedef __attribute__((ext_vector_type(8)))  float v8f;

#define NB    2
#define CCH   64
#define DDEP  32
#define HH    64
#define WWID  64
#define SPN   131072   /* D*H*W spatial per batch */
#define NHEAD 4
#define HD    16

union FragU { v16bf v; uint4 u[2]; };

__device__ __forceinline__ v8f wmma_bf16(const FragU& a, const FragU& b, v8f c) {
  return __builtin_amdgcn_wmma_f32_16x16x32_bf16(false, a.v, false, b.v,
                                                 (short)0, c, false, false);
}

// ---------------------------------------------------------------------------
// CDNA5 async global->LDS copy (16B per lane) + completion wait.
// lds_off is a byte offset into the workgroup's dynamic LDS allocation
// (kernels using this keep ALL shared memory dynamic so the base is 0).
// ---------------------------------------------------------------------------
__device__ __forceinline__ void async_cp16(uint32_t lds_off, const void* g) {
  asm volatile("global_load_async_to_lds_b128 %0, %1, off"
               :: "v"(lds_off), "v"((uint64_t)(size_t)g) : "memory");
}
__device__ __forceinline__ void async_wait0() {
  asm volatile("s_wait_asynccnt 0x0" ::: "memory");
}

// ---------------------------------------------------------------------------
// Tensor Data Mover: one instruction DMAs a 16(z) x 32(d) x 64(y) bf16 tile
// (64KB) from global to LDS.  D# built from wave-uniform scalars.
// ---------------------------------------------------------------------------
#if __has_builtin(__builtin_amdgcn_tensor_load_to_lds)
#define HAVE_TDM 1
typedef unsigned int v4u __attribute__((ext_vector_type(4)));
typedef int v4i __attribute__((ext_vector_type(4)));
typedef int v8i __attribute__((ext_vector_type(8)));

__device__ __forceinline__ void tdm_load_slab(uint32_t lds_off, const void* gp) {
  uint64_t ga = (uint64_t)(size_t)gp;
  // group0: count=1 | lds_addr | global_addr | type=2 (bits 127:126)
  v4u g0 = { 1u, lds_off, (uint32_t)ga, (uint32_t)(ga >> 32) | 0x80000000u };
  // group1: data_size=2B; tensor_dim0=64, tensor_dim1=32;
  //         tile_dim0=64, tile_dim1=32, tile_dim2=16;
  //         dim0_stride=4096 elems, dim1_stride=131072 elems
  v8i g1 = { 0x00010000, 0x00400000, 0x00200000, 0x00400000,
             0x00100020, 0x00001000, 0x00000000, 0x00000002 };
  // group2: tensor_dim2=16; no dim3; group3: zeros (3D tile)
  v4i g2 = { 16, 1, 0, 0 };
  v4i g3 = { 0, 0, 0, 0 };
#if defined(__clang_major__) && (__clang_major__ >= 23)
  v8i g4 = { 0, 0, 0, 0, 0, 0, 0, 0 };
  __builtin_amdgcn_tensor_load_to_lds(g0, g1, g2, g3, g4, 0);
#else
  __builtin_amdgcn_tensor_load_to_lds(g0, g1, g2, g3, 0);
#endif
}
#endif

__device__ __forceinline__ void tensor_wait0() {
#if __has_builtin(__builtin_amdgcn_s_wait_tensorcnt)
  __builtin_amdgcn_s_wait_tensorcnt(0);
#else
  asm volatile("s_wait_tensorcnt 0x0" ::: "memory");
#endif
}

// ---------------------------------------------------------------------------
// Kernel 0: zero BN stats accumulators
// ---------------------------------------------------------------------------
__global__ void zero_stats_kernel(float* __restrict__ stats) {
  if (threadIdx.x < 128) stats[threadIdx.x] = 0.0f;
}

// ---------------------------------------------------------------------------
// Kernel 1: fused Q/K/V projection.  out[o,s] = sum_c W[o,c] * x[c,s]
// M=64, K=64, N tile = 128 spatial columns per workgroup.  bf16 WMMA.
// ---------------------------------------------------------------------------
#define K1_WL 0
#define K1_XT 27648
#define K1_XS 46080
#define K1_SMEM 78848

__global__ __launch_bounds__(256) void qkv_proj_kernel(
    const float* __restrict__ x,  const float* __restrict__ Wq,
    const float* __restrict__ Wk, const float* __restrict__ Wv,
    bf16* __restrict__ qb, bf16* __restrict__ kbuf, bf16* __restrict__ vbuf)
{
  extern __shared__ char smem[];
  bf16* wl = (bf16*)(smem + K1_WL);
  bf16* xt = (bf16*)(smem + K1_XT);
  const int t   = threadIdx.x;
  const int bid = blockIdx.x;
  const int n   = bid >> 10;             // 1024 blocks per batch
  const int s0  = (bid & 1023) << 7;     // *128
  const float* Ws[3] = {Wq, Wk, Wv};

  // async stage the raw fp32 x tile: 64 rows x 512B -> 2048 16B chunks
  for (int i = 0; i < 8; ++i) {
    int idx = t + i * 256;
    int c = idx >> 5, ch = idx & 31;
    async_cp16(K1_XS + idx * 16,
               x + (size_t)(n * CCH + c) * SPN + s0 + ch * 4);
  }
  // overlap: load + convert W while the async engine streams x
  for (int i = 0; i < 48; ++i) {         // 3*64*64 = 12288 W elems
    int e = t + i * 256;
    int mat = e >> 12, rem = e & 4095;
    int o = rem >> 6, c = rem & 63;
    wl[mat * 64 * 72 + o * 72 + c] = (bf16)Ws[mat][o * 64 + c];
  }
  async_wait0();
  __syncthreads();
  const float* xs = (const float*)(smem + K1_XS);
  for (int i = 0; i < 32; ++i) {         // transpose + convert LDS->LDS
    int e = t + i * 256;
    int c = e >> 7, j = e & 127;
    xt[j * 72 + c] = (bf16)xs[c * 128 + j];
  }
  __syncthreads();

  const int wv = t >> 5, l = t & 31, lm = l & 15, hi = l >> 4;
  bf16* outs[3] = {qb, kbuf, vbuf};
  for (int ji = 0; ji < 4; ++ji) {
    int job   = wv + (ji << 3);          // 32 jobs: 4 o-tiles x 8 strips
    int otile = job & 3, strip = job >> 2;
    v8f acc[3] = {v8f{}, v8f{}, v8f{}};
    for (int ks = 0; ks < 2; ++ks) {     // K = 64 -> two 32-chunks
      FragU bfr;
      const char* bb = (const char*)xt + (strip * 16 + lm) * 144 + (ks * 32 + hi * 16) * 2;
      bfr.u[0] = *(const uint4*)bb;
      bfr.u[1] = *(const uint4*)(bb + 16);
      for (int m = 0; m < 3; ++m) {
        FragU af;
        const char* ab = (const char*)(wl + m * 64 * 72) + (otile * 16 + lm) * 144
                       + (ks * 32 + hi * 16) * 2;
        af.u[0] = *(const uint4*)ab;
        af.u[1] = *(const uint4*)(ab + 16);
        acc[m] = wmma_bf16(af, bfr, acc[m]);
      }
    }
    int sidx = s0 + strip * 16 + lm;
    for (int m = 0; m < 3; ++m) {
      bf16* op = outs[m] + (size_t)(n * CCH) * SPN + sidx;
      for (int r = 0; r < 8; ++r) {
        int o = otile * 16 + r + hi * 8;
        op[(size_t)o * SPN] = (bf16)acc[m][r];
      }
    }
  }
}

// ---------------------------------------------------------------------------
// Kernel 2: fused attention per (n, head, x) slab.
// Dynamic LDS layout (bytes):
//   QO      0 (66560) : qT[y][d][z] then vT[y][z][d]   (y stride 1040)
//   KO  66560 (73728) : kT[y][e][z] then out[z*32+e][y] (row stride 144)
//                       also v raw TDM stage (65536) after energy
//   AO 140288 (132096): att[y][e][d] bf16 (y stride 2064)
//                       first reused as q (AO) / k (AO+65536) raw TDM stages
//   ZO 272384   (128) : zeros for WMMA K-padding lanes
// ---------------------------------------------------------------------------
#define QO 0
#define KO 66560
#define AO 140288
#define ZO 272384
#define SMEM2_BYTES 272512

__global__ __launch_bounds__(512) void attn_kernel(
    const bf16* qb, const bf16* kb, const bf16* vb, bf16* attn /* aliases qb */)
{
  extern __shared__ char smem[];
  const int t  = threadIdx.x, wv = t >> 5, l = t & 31;
  const int lm = l & 15, hi = l >> 4;
  const int bid = blockIdx.x;
  const int n  = bid >> 8;
  const int h  = (bid >> 6) & 3;
  const int xr = bid & 63;
  const size_t base = ((size_t)(n * CCH + h * HD)) * SPN + (size_t)xr * WWID;
  const bf16* qg = qb + base;
  const bf16* kg = kb + base;
  const bf16* vg = vb + base;

  // ---- stage raw q,k slabs (16z x 32d x 64y bf16 tiles) ----
#if defined(HAVE_TDM)
  if (wv == 0) {
    tdm_load_slab(AO, qg);
    tdm_load_slab(AO + 65536, kg);
  }
#else
  for (int i = 0; i < 8; ++i) {
    int idx = t + i * 512;
    int rr = idx >> 3, ch = idx & 7;
    int z = rr >> 5, d = rr & 31;
    async_cp16(AO + rr * 128 + ch * 16, qg + (size_t)z * SPN + d * 4096 + ch * 8);
    async_cp16(AO + 65536 + rr * 128 + ch * 16, kg + (size_t)z * SPN + d * 4096 + ch * 8);
  }
#endif
  if (t < 32) *(float*)(smem + ZO + t * 4) = 0.0f;
  // warm L2 for the v slab while q/k stream + energy runs
  __builtin_prefetch(vg + (size_t)(t >> 5) * SPN + (size_t)(t & 31) * 4096, 0, 0);
#if defined(HAVE_TDM)
  if (wv == 0) tensor_wait0();
#else
  async_wait0();
#endif
  __syncthreads();

  // ---- transpose stages: qT[y][d][z], kT[y][e][z] ----
  for (int rr = wv; rr < 512; rr += 16) {
    int z = rr >> 5, d = rr & 31;
    uint32_t tq = *(const uint32_t*)(smem + AO + rr * 128 + l * 4);
    uint32_t tk = *(const uint32_t*)(smem + AO + 65536 + rr * 128 + l * 4);
    bf16 q0 = ((const bf16*)&tq)[0], q1 = ((const bf16*)&tq)[1];
    bf16 k0 = ((const bf16*)&tk)[0], k1 = ((const bf16*)&tk)[1];
    int y0 = 2 * l, y1 = 2 * l + 1;
    *(bf16*)(smem + QO + y0 * 1040 + d * 32 + z * 2) = q0;
    *(bf16*)(smem + QO + y1 * 1040 + d * 32 + z * 2) = q1;
    *(bf16*)(smem + KO + y0 * 1040 + d * 32 + z * 2) = k0;  // d acts as e
    *(bf16*)(smem + KO + y1 * 1040 + d * 32 + z * 2) = k1;
  }
  __syncthreads();

  // ---- energy: E[e,d] = sum_z K[z,e]*Q[z,d], K=16 padded to 32 ----
  for (int y = wv; y < 64; y += 16) {
    for (int dt = 0; dt < 2; ++dt) {
      FragU bq;  // B[k=z][n=d]; VGPR4-7 carry K=16..31 -> zeros
      const char* bp = smem + QO + y * 1040 + (dt * 16 + lm) * 32 + hi * 16;
      bq.u[0] = *(const uint4*)bp;
      bq.u[1] = make_uint4(0u, 0u, 0u, 0u);
      for (int et = 0; et < 2; ++et) {
        FragU ak;  // A[m=e][k=z]; VGPR4-7 carry K=16..31 -> zeros
        const char* ap = smem + KO + y * 1040 + (et * 16 + lm) * 32 + hi * 16;
        ak.u[0] = *(const uint4*)ap;
        ak.u[1] = make_uint4(0u, 0u, 0u, 0u);
        v8f e = {};
        e = wmma_bf16(ak, bq, e);
        for (int r = 0; r < 8; ++r) {
          int ea = et * 16 + r + hi * 8;
          int da = dt * 16 + lm;
          *(bf16*)(smem + AO + y * 2064 + ea * 64 + da * 2) = (bf16)e[r];
        }
      }
    }
  }
  __syncthreads();

  // ---- kT is dead: DMA raw v slab into KO while softmax runs ----
#if defined(HAVE_TDM)
  if (wv == 0) tdm_load_slab(KO, vg);
#else
  for (int i = 0; i < 8; ++i) {
    int idx = t + i * 512;
    int rr = idx >> 3, ch = idx & 7;
    int z = rr >> 5, d = rr & 31;
    async_cp16(KO + rr * 128 + ch * 16, vg + (size_t)z * SPN + d * 4096 + ch * 8);
  }
#endif

  // ---- softmax over y per (e,d) row ----
  for (int p = t; p < 1024; p += 512) {
    int e = p >> 5, d = p & 31;
    char* rowp = smem + AO + e * 64 + d * 2;
    float mx = -1e30f;
    for (int y = 0; y < 64; ++y)
      mx = fmaxf(mx, (float)*(bf16*)(rowp + y * 2064));
    float s = 0.0f;
    for (int y = 0; y < 64; ++y)
      s += __expf((float)*(bf16*)(rowp + y * 2064) - mx);
    float inv = 1.0f / s;
    for (int y = 0; y < 64; ++y) {
      float v = (float)*(bf16*)(rowp + y * 2064);
      *(bf16*)(rowp + y * 2064) = (bf16)(__expf(v - mx) * inv);
    }
  }
#if defined(HAVE_TDM)
  if (wv == 0) tensor_wait0();
#else
  async_wait0();
#endif
  __syncthreads();

  // ---- transpose v stage into QO region as vT[y][z][d] (q is dead) ----
  for (int rr = wv; rr < 512; rr += 16) {
    int z = rr >> 5, d = rr & 31;
    uint32_t tv = *(const uint32_t*)(smem + KO + rr * 128 + l * 4);
    bf16 v0 = ((const bf16*)&tv)[0], v1 = ((const bf16*)&tv)[1];
    *(bf16*)(smem + QO + (2 * l)     * 1040 + z * 64 + d * 2) = v0;
    *(bf16*)(smem + QO + (2 * l + 1) * 1040 + z * 64 + d * 2) = v1;
  }
  __syncthreads();

  // ---- AV: out[z,e] = sum_d V[z,d]*att[e,d], full K=32; out into KO region ----
  for (int y = wv; y < 64; y += 16) {
    FragU av;  // A[m=z][k=d]
    const char* ap = smem + QO + y * 1040 + lm * 64 + hi * 32;
    av.u[0] = *(const uint4*)ap;
    av.u[1] = *(const uint4*)(ap + 16);
    for (int et = 0; et < 2; ++et) {
      FragU ba;  // B[k=d][n=e]
      const char* bp = smem + AO + y * 2064 + (et * 16 + lm) * 64 + hi * 32;
      ba.u[0] = *(const uint4*)bp;
      ba.u[1] = *(const uint4*)(bp + 16);
      v8f o = {};
      o = wmma_bf16(av, ba, o);
      for (int r = 0; r < 8; ++r) {
        int z  = r + hi * 8;
        int ea = et * 16 + lm;
        *(bf16*)(smem + KO + (z * 32 + ea) * 144 + y * 2) = (bf16)o[r];
      }
    }
  }
  __syncthreads();

  // ---- coalesced slab store: attn[n, h*16+z, e, x, y] ----
  bf16* og = attn + base;
  for (int rr = wv; rr < 512; rr += 16) {
    int z = rr >> 5, e = rr & 31;
    uint32_t val = *(const uint32_t*)(smem + KO + rr * 144 + l * 4);
    *(uint32_t*)(og + (size_t)z * SPN + e * 4096 + 2 * l) = val;
  }
}

// ---------------------------------------------------------------------------
// Kernel 3: fc_out GEMM + bias -> fp32 y, plus per-channel BN sum/sumsq
// ---------------------------------------------------------------------------
#define K3_WL 0
#define K3_XT 9216
#define K3_SUM 27648
#define K3_SSQ 27904
#define K3_AS 28160
#define K3_SMEM 44544

__global__ __launch_bounds__(256) void fc_bn_stats_kernel(
    const bf16* __restrict__ attn, const float* __restrict__ Wo,
    const float* __restrict__ bo,  float* __restrict__ ybuf,
    float* __restrict__ stats)
{
  extern __shared__ char smem[];
  bf16* wl = (bf16*)(smem + K3_WL);
  bf16* xt = (bf16*)(smem + K3_XT);
  float* ssum = (float*)(smem + K3_SUM);
  float* ssq  = (float*)(smem + K3_SSQ);
  const int t = threadIdx.x;
  const int bid = blockIdx.x;
  const int n = bid >> 10, s0 = (bid & 1023) << 7;

  // async stage the raw bf16 attn tile: 64 rows x 256B -> 1024 16B chunks
  for (int i = 0; i < 4; ++i) {
    int idx = t + i * 256;
    int c = idx >> 4, ch = idx & 15;
    async_cp16(K3_AS + idx * 16,
               attn + (size_t)(n * CCH + c) * SPN + s0 + ch * 8);
  }
  for (int i = 0; i < 16; ++i) {
    int e = t + i * 256;
    wl[(e >> 6) * 72 + (e & 63)] = (bf16)Wo[e];
  }
  if (t < 64) { ssum[t] = 0.0f; ssq[t] = 0.0f; }
  async_wait0();
  __syncthreads();
  const bf16* as = (const bf16*)(smem + K3_AS);
  for (int i = 0; i < 32; ++i) {         // transpose 8192 bf16 LDS->LDS
    int e = t + i * 256;
    int c = e >> 7, j = e & 127;
    xt[j * 72 + c] = as[c * 128 + j];
  }
  __syncthreads();

  const int wv = t >> 5, l = t & 31, lm = l & 15, hi = l >> 4;
  for (int ji = 0; ji < 4; ++ji) {
    int job   = wv + (ji << 3);
    int otile = job & 3, strip = job >> 2;
    v8f acc = {};
    for (int ks = 0; ks < 2; ++ks) {
      FragU bfr, af;
      const char* bb = (const char*)xt + (strip * 16 + lm) * 144 + (ks * 32 + hi * 16) * 2;
      bfr.u[0] = *(const uint4*)bb;
      bfr.u[1] = *(const uint4*)(bb + 16);
      const char* ab = (const char*)wl + (otile * 16 + lm) * 144 + (ks * 32 + hi * 16) * 2;
      af.u[0] = *(const uint4*)ab;
      af.u[1] = *(const uint4*)(ab + 16);
      acc = wmma_bf16(af, bfr, acc);
    }
    int sidx = s0 + strip * 16 + lm;
    for (int r = 0; r < 8; ++r) {
      int o = otile * 16 + r + hi * 8;
      float vout = acc[r] + bo[o];
      ybuf[(size_t)(n * CCH + o) * SPN + sidx] = vout;
      atomicAdd(&ssum[o], vout);
      atomicAdd(&ssq[o],  vout * vout);
    }
  }
  __syncthreads();
  if (t < 64) {
    atomicAdd(&stats[t],      ssum[t]);
    atomicAdd(&stats[64 + t], ssq[t]);
  }
}

// ---------------------------------------------------------------------------
// Kernel 4: BN (batch stats) + relu + residual, float4 vectorized
// ---------------------------------------------------------------------------
__global__ __launch_bounds__(256) void bn_apply_kernel(
    const float* __restrict__ ybuf, const float* __restrict__ x,
    const float* __restrict__ stats, const float* __restrict__ gamma,
    const float* __restrict__ beta, float* __restrict__ out)
{
  const size_t i = ((size_t)blockIdx.x * 256 + threadIdx.x) * 4;
  const int c = (int)((i >> 17) & 63);
  const float cnt = 262144.0f;           // N*D*H*W
  float mean = stats[c] / cnt;
  float var  = stats[64 + c] / cnt - mean * mean;
  float inv  = rsqrtf(var + 1e-5f) * gamma[c];
  float sh   = beta[c] - mean * inv;
  float4 yv = *(const float4*)(ybuf + i);
  float4 xv = *(const float4*)(x + i);
  float4 r;
  r.x = fmaxf(yv.x * inv + sh, 0.0f) + xv.x;
  r.y = fmaxf(yv.y * inv + sh, 0.0f) + xv.y;
  r.z = fmaxf(yv.z * inv + sh, 0.0f) + xv.z;
  r.w = fmaxf(yv.w * inv + sh, 0.0f) + xv.w;
  *(float4*)(out + i) = r;
}

// ---------------------------------------------------------------------------
extern "C" void kernel_launch(void* const* d_in, const int* in_sizes, int n_in,
                              void* d_out, int out_size, void* d_ws, size_t ws_size,
                              hipStream_t stream) {
  (void)in_sizes; (void)n_in; (void)out_size; (void)ws_size;
  const float* x     = (const float*)d_in[0];
  const float* Wq    = (const float*)d_in[1];
  const float* Wk    = (const float*)d_in[2];
  const float* Wv    = (const float*)d_in[3];
  const float* Wo    = (const float*)d_in[4];
  const float* bo    = (const float*)d_in[5];
  const float* gamma = (const float*)d_in[6];
  const float* beta  = (const float*)d_in[7];

  char* ws = (char*)d_ws;
  bf16*  qb    = (bf16*)ws;                    // 64 MiB; later aliased as attn_out
  bf16*  kbuf  = (bf16*)(ws + 67108864);       // 64 MiB
  bf16*  vbuf  = (bf16*)(ws + 134217728);      // 64 MiB
  float* ybuf  = (float*)(ws + 67108864);      // 128 MiB, overwrites k+v after attn
  float* stats = (float*)(ws + 201326592);     // 128 floats
  float* out   = (float*)d_out;

  zero_stats_kernel<<<1, 128, 0, stream>>>(stats);
  qkv_proj_kernel<<<2048, 256, K1_SMEM, stream>>>(x, Wq, Wk, Wv, qb, kbuf, vbuf);
  attn_kernel<<<512, 512, SMEM2_BYTES, stream>>>(qb, kbuf, vbuf, qb);
  fc_bn_stats_kernel<<<2048, 256, K3_SMEM, stream>>>(qb, Wo, bo, ybuf, stats);
  bn_apply_kernel<<<32768, 256, 0, stream>>>(ybuf, x, stats, gamma, beta, out);
}
